// AgentAttention_41326175322318
// MI455X (gfx1250) — matrix operand order repaired and checked by reading, not compiled, not run on hardware
//
#include <hip/hip_runtime.h>
#include <hip/hip_bf16.h>

typedef __bf16 bf16_t;
typedef __attribute__((ext_vector_type(16))) __bf16 v16bf;
typedef __attribute__((ext_vector_type(8)))  __bf16 v8bf;
typedef __attribute__((ext_vector_type(8)))  float  v8f;

#define BB 8
#define NN 4096
#define SS 4096
#define DD 1024
#define HH 16
#define AA 8
#define HD 64
#define SCALE 0.125f
#define KKEEP 2048
#define MBIG (BB * NN)          // 32768
#define MAGENT 128              // padded (real rows = B*A = 64)

// ---------------------------------------------------------------------------
// utility kernels
// ---------------------------------------------------------------------------
__global__ void zero_f32_kernel(float* __restrict__ p, int n) {
    int i = blockIdx.x * 256 + threadIdx.x;
    if (i < n) p[i] = 0.0f;
}

__global__ void f32_to_bf16_kernel(const float* __restrict__ src,
                                   bf16_t* __restrict__ dst, int n) {
    int i = blockIdx.x * 256 + threadIdx.x;
    if (i < n) dst[i] = (bf16_t)src[i];
}

// Pack a 1024x1024 row-major f32 weight into WMMA B-fragment order (bf16).
// Layout: Wp[((kb*64 + nb)*32 + lane)*16 + i]
//   lane 0-15 : n = nb*16 + lane,      element i -> K = kb*32 + i
//   lane 16-31: n = nb*16 + (lane-16), element i -> K = kb*32 + 16 + i
__global__ void pack_w_kernel(const float* __restrict__ W,
                              bf16_t* __restrict__ Wp) {
    int idx = blockIdx.x * 256 + threadIdx.x;      // 0 .. 1024*1024-1
    int i    = idx & 15;
    int lane = (idx >> 4) & 31;
    int nb   = (idx >> 9) & 63;
    int kb   = idx >> 15;
    int k = kb * 32 + ((lane >= 16) ? 16 : 0) + i;
    int n = nb * 16 + (lane & 15);
    Wp[idx] = (bf16_t)W[(size_t)k * DD + n];
}

// agent_mean (padded to 128 rows; rows >= 64 get 0)
__global__ void agent_mean_kernel(const float* __restrict__ q,
                                  float* __restrict__ out) {
    int idx = blockIdx.x * 256 + threadIdx.x;   // 0 .. 128*1024-1
    int col = idx & (DD - 1);
    int row = idx >> 10;
    float s = 0.0f;
    if (row < BB * AA) {
        int b = row >> 3, a = row & 7;
        const float* p = q + ((size_t)b * NN + (size_t)a * (NN / AA)) * DD + col;
        for (int i = 0; i < NN / AA; ++i) s += p[(size_t)i * DD];
        s *= (1.0f / (float)(NN / AA));
    }
    out[idx] = s;
}

// ---------------------------------------------------------------------------
// bf16 WMMA GEMM:  C[M,1024] = A[M,1024](bf16,row-major) * Wp(packed) + bias
// block = 256 threads (8 waves), tile 128(M) x 64(N); wave tile 32x32.
// ---------------------------------------------------------------------------
__global__ __launch_bounds__(256)
void gemm_bf16_kernel(const bf16_t* __restrict__ A,
                      const bf16_t* __restrict__ Wp,
                      const float* __restrict__ bias,
                      float* __restrict__ C, int M) {
    int lane = threadIdx.x & 31;
    int wave = threadIdx.x >> 5;            // 0..7
    int wm   = (wave >> 1) * 32;            // 0,32,64,96
    int wn   = (wave & 1) * 32;             // 0,32
    int m0   = blockIdx.y * 128 + wm;
    int n0   = blockIdx.x * 64 + wn;

    int mrow  = lane & 15;
    int koffA = (lane < 16) ? 0 : 16;       // A-frag K base per half-wave
    const bf16_t* Arow0 = A + (size_t)(m0 + mrow) * DD;
    const bf16_t* Arow1 = A + (size_t)(m0 + 16 + mrow) * DD;
    int nb0 = n0 >> 4;

    v8f acc00 = {}, acc01 = {}, acc10 = {}, acc11 = {};

    for (int kb = 0; kb < DD / 32; ++kb) {
        int k0 = kb * 32;
        union { v16bf v; v8bf h[2]; } a0, a1, b0, b1;
        a0.h[0] = *(const v8bf*)(Arow0 + k0 + koffA);
        a0.h[1] = *(const v8bf*)(Arow0 + k0 + koffA + 8);
        a1.h[0] = *(const v8bf*)(Arow1 + k0 + koffA);
        a1.h[1] = *(const v8bf*)(Arow1 + k0 + koffA + 8);
        const bf16_t* bp0 = Wp + ((size_t)(kb * 64 + nb0) * 32 + lane) * 16;
        const bf16_t* bp1 = Wp + ((size_t)(kb * 64 + nb0 + 1) * 32 + lane) * 16;
        b0.h[0] = *(const v8bf*)(bp0);
        b0.h[1] = *(const v8bf*)(bp0 + 8);
        b1.h[0] = *(const v8bf*)(bp1);
        b1.h[1] = *(const v8bf*)(bp1 + 8);

        acc00 = __builtin_amdgcn_wmma_f32_16x16x32_bf16(false, a0.v, false, b0.v, (short)0, acc00, false, false);
        acc01 = __builtin_amdgcn_wmma_f32_16x16x32_bf16(false, a0.v, false, b1.v, (short)0, acc01, false, false);
        acc10 = __builtin_amdgcn_wmma_f32_16x16x32_bf16(false, a1.v, false, b0.v, (short)0, acc10, false, false);
        acc11 = __builtin_amdgcn_wmma_f32_16x16x32_bf16(false, a1.v, false, b1.v, (short)0, acc11, false, false);
    }

    int col0 = n0 + (lane & 15);
    int col1 = col0 + 16;
    float bv0 = bias[col0];
    float bv1 = bias[col1];
    int rbase = (lane < 16) ? 0 : 8;
#pragma unroll
    for (int r = 0; r < 8; ++r) {
        int row0 = m0 + rbase + r;
        int row1 = row0 + 16;
        C[(size_t)row0 * DD + col0] = acc00[r] + bv0;
        C[(size_t)row0 * DD + col1] = acc01[r] + bv1;
        C[(size_t)row1 * DD + col0] = acc10[r] + bv0;
        C[(size_t)row1 * DD + col1] = acc11[r] + bv1;
    }
}

// ---------------------------------------------------------------------------
// agent attention scores: scores[bh][a][s] = SCALE * dot64(q_a, k_c)
// ---------------------------------------------------------------------------
__global__ __launch_bounds__(256)
void agent_score_kernel(const float* __restrict__ qa,
                        const float* __restrict__ kc,
                        float* __restrict__ scores) {
    __shared__ float qs[AA * HD];
    int bh = blockIdx.y;                     // 0..127
    int b = bh >> 4, h = bh & 15;
    int s = blockIdx.x * 256 + threadIdx.x;
    for (int i = threadIdx.x; i < AA * HD; i += 256)
        qs[i] = qa[(size_t)(b * AA + (i >> 6)) * DD + h * HD + (i & 63)];
    __syncthreads();

    const float* kp = kc + ((size_t)(b * SS + s)) * DD + h * HD;
    float k[HD];
#pragma unroll
    for (int i = 0; i < 16; ++i) {
        float4 t = ((const float4*)kp)[i];
        k[4 * i] = t.x; k[4 * i + 1] = t.y; k[4 * i + 2] = t.z; k[4 * i + 3] = t.w;
    }
#pragma unroll
    for (int a = 0; a < AA; ++a) {
        float d = 0.0f;
#pragma unroll
        for (int i = 0; i < HD; ++i) d += k[i] * qs[a * HD + i];
        scores[((size_t)bh * AA + a) * SS + s] = d * SCALE;
    }
}

// ---------------------------------------------------------------------------
// exact top-k (k=2048) threshold via MSB-first radix select + masked softmax
// one 256-thread workgroup per row of 4096 scores (in-place in d_out slice)
// ---------------------------------------------------------------------------
__device__ __forceinline__ float key_to_float(unsigned key) {
    unsigned u = (key & 0x80000000u) ? (key & 0x7FFFFFFFu) : ~key;
    return __uint_as_float(u);
}

__global__ __launch_bounds__(256)
void topk_softmax_kernel(float* __restrict__ scores) {
    __shared__ unsigned keys[SS];
    __shared__ unsigned hist[256];
    __shared__ unsigned red[256];
    __shared__ unsigned s_prefix, s_rank;

    int tid = threadIdx.x;
    float* p = scores + (size_t)blockIdx.x * SS;

    unsigned kmax = 0;
    for (int i = tid; i < SS; i += 256) {
        unsigned u = __float_as_uint(p[i]);
        unsigned key = (u & 0x80000000u) ? ~u : (u | 0x80000000u);
        keys[i] = key;
        kmax = (key > kmax) ? key : kmax;
    }
    red[tid] = kmax;
    __syncthreads();
    for (int off = 128; off; off >>= 1) {
        if (tid < off) { unsigned o = red[tid + off]; if (o > red[tid]) red[tid] = o; }
        __syncthreads();
    }
    float rowmax = key_to_float(red[0]);

    if (tid == 0) { s_prefix = 0u; s_rank = KKEEP; }
    __syncthreads();
    for (int pass = 0; pass < 4; ++pass) {
        int shift = 24 - pass * 8;
        hist[tid] = 0u;
        __syncthreads();
        unsigned prefix = s_prefix;
        unsigned pmask = (pass == 0) ? 0u : (0xFFFFFFFFu << (shift + 8));
        for (int i = tid; i < SS; i += 256) {
            unsigned key = keys[i];
            if ((key & pmask) == prefix)
                atomicAdd(&hist[(key >> shift) & 255u], 1u);
        }
        __syncthreads();
        if (tid == 0) {
            unsigned r = s_rank, cum = 0; int bsel = 0;
            for (int bck = 255; bck >= 0; --bck) {
                unsigned c = hist[bck];
                if (cum + c >= r) { bsel = bck; s_rank = r - cum; break; }
                cum += c;
            }
            s_prefix = prefix | ((unsigned)bsel << shift);
        }
        __syncthreads();
    }
    unsigned tkey = s_prefix;      // key of the k-th largest value

    float lsum = 0.0f;
    for (int i = tid; i < SS; i += 256) {
        unsigned key = keys[i];
        float e = 0.0f;
        if (key >= tkey) e = __expf(key_to_float(key) - rowmax);
        keys[i] = __float_as_uint(e);
        lsum += e;
    }
    ((float*)red)[tid] = lsum;
    __syncthreads();
    for (int off = 128; off; off >>= 1) {
        if (tid < off) ((float*)red)[tid] += ((float*)red)[tid + off];
        __syncthreads();
    }
    float inv = 1.0f / ((float*)red)[0];
    for (int i = tid; i < SS; i += 256)
        p[i] = __uint_as_float(keys[i]) * inv;
}

// ---------------------------------------------------------------------------
// agent_out[b][a][h*64+d] = sum_s w[bh][a][s] * v_c[b][s][h*64+d]
// ---------------------------------------------------------------------------
#define TSZ 128
__global__ __launch_bounds__(256)
void agent_out_kernel(const float* __restrict__ w,
                      const float* __restrict__ vc,
                      float* __restrict__ aout) {
    __shared__ float vt[TSZ * HD];
    __shared__ float wt[AA * TSZ];
    int bh = blockIdx.x;
    int b = bh >> 4, h = bh & 15;
    int tid = threadIdx.x;
    int o0 = tid, o1 = tid + 256;
    int a0 = o0 >> 6, d0 = o0 & 63;
    int a1 = o1 >> 6, d1 = o1 & 63;
    float acc0 = 0.0f, acc1 = 0.0f;

    for (int s0 = 0; s0 < SS; s0 += TSZ) {
        __syncthreads();
        for (int i = tid; i < TSZ * HD; i += 256) {
            int ss = i >> 6, d = i & 63;
            vt[i] = vc[((size_t)(b * SS + s0 + ss)) * DD + h * HD + d];
        }
        for (int i = tid; i < AA * TSZ; i += 256) {
            int a = i / TSZ, ss = i % TSZ;
            wt[i] = w[((size_t)bh * AA + a) * SS + s0 + ss];
        }
        __syncthreads();
#pragma unroll 8
        for (int ss = 0; ss < TSZ; ++ss) {
            acc0 += wt[a0 * TSZ + ss] * vt[ss * HD + d0];
            acc1 += wt[a1 * TSZ + ss] * vt[ss * HD + d1];
        }
    }
    aout[((size_t)(b * AA + a0)) * DD + h * HD + d0] = acc0;
    aout[((size_t)(b * AA + a1)) * DD + h * HD + d1] = acc1;
}

// ---------------------------------------------------------------------------
// final attention: per (b,h,n): softmax over A=8, write attn_final + bf16 ctx
// ---------------------------------------------------------------------------
__global__ __launch_bounds__(256)
void final_attn_kernel(const float* __restrict__ qo,
                       const float* __restrict__ ka,
                       const float* __restrict__ va,
                       float* __restrict__ attn_out,
                       bf16_t* __restrict__ ctx) {
    __shared__ float kl[AA * HD];
    __shared__ float vl[AA * HD];
    int bh = blockIdx.y;
    int b = bh >> 4, h = bh & 15;
    int n = blockIdx.x * 256 + threadIdx.x;
    for (int i = threadIdx.x; i < AA * HD; i += 256) {
        int a = i >> 6, d = i & 63;
        kl[i] = ka[((size_t)(b * AA + a)) * DD + h * HD + d];
        vl[i] = va[((size_t)(b * AA + a)) * DD + h * HD + d];
    }
    __syncthreads();

    const float* qp = qo + ((size_t)(b * NN + n)) * DD + h * HD;
    float q[HD];
#pragma unroll
    for (int i = 0; i < 16; ++i) {
        float4 t = ((const float4*)qp)[i];
        q[4 * i] = t.x; q[4 * i + 1] = t.y; q[4 * i + 2] = t.z; q[4 * i + 3] = t.w;
    }
    float sc[AA];
    float m = -3.402823466e+38f;
#pragma unroll
    for (int a = 0; a < AA; ++a) {
        float d = 0.0f;
#pragma unroll
        for (int i = 0; i < HD; ++i) d += q[i] * kl[a * HD + i];
        sc[a] = d * SCALE;
        m = fmaxf(m, sc[a]);
    }
    float sum = 0.0f;
#pragma unroll
    for (int a = 0; a < AA; ++a) { sc[a] = __expf(sc[a] - m); sum += sc[a]; }
    float inv = 1.0f / sum;
#pragma unroll
    for (int a = 0; a < AA; ++a) sc[a] *= inv;

    float* ap = attn_out + ((size_t)bh * NN + n) * AA;
    float4 w0 = make_float4(sc[0], sc[1], sc[2], sc[3]);
    float4 w1 = make_float4(sc[4], sc[5], sc[6], sc[7]);
    ((float4*)ap)[0] = w0;
    ((float4*)ap)[1] = w1;

    bf16_t* cp = ctx + ((size_t)(b * NN + n)) * DD + h * HD;
#pragma unroll
    for (int dblk = 0; dblk < 8; ++dblk) {
        v8bf o;
#pragma unroll
        for (int j = 0; j < 8; ++j) {
            int d = dblk * 8 + j;
            float acc = 0.0f;
#pragma unroll
            for (int a = 0; a < AA; ++a) acc += sc[a] * vl[a * HD + d];
            o[j] = (bf16_t)acc;
        }
        *(v8bf*)(cp + dblk * 8) = o;
    }
}

// ---------------------------------------------------------------------------
// host launcher
// ---------------------------------------------------------------------------
extern "C" void kernel_launch(void* const* d_in, const int* in_sizes, int n_in,
                              void* d_out, int out_size, void* d_ws, size_t ws_size,
                              hipStream_t stream) {
    const float* query = (const float*)d_in[0];
    const float* key   = (const float*)d_in[1];
    const float* value = (const float*)d_in[2];
    // weight/bias order: agent, qa, kc, vc, qo, ka, va, proj
    const float* Wf[8];  const float* bf[8];
    for (int i = 0; i < 8; ++i) { Wf[i] = (const float*)d_in[3 + 2 * i]; bf[i] = (const float*)d_in[4 + 2 * i]; }

    float* out_final  = (float*)d_out;                               // B*N*D
    float* attn_final = out_final + (size_t)BB * NN * DD;            // B*H*N*A
    float* attn_agent = attn_final + (size_t)BB * HH * NN * AA;      // B*H*A*S

    // workspace carve-out (all sizes multiples of 256B)
    char* w = (char*)d_ws;
    size_t off = 0;
    auto alloc = [&](size_t bytes) -> void* {
        void* p = w + off; off += (bytes + 255) & ~(size_t)255; return p;
    };
    bf16_t* q_bf   = (bf16_t*)alloc((size_t)MBIG * DD * 2);
    bf16_t* k_bf   = (bf16_t*)alloc((size_t)MBIG * DD * 2);
    bf16_t* v_bf   = (bf16_t*)alloc((size_t)MBIG * DD * 2);
    float*  kc_f   = (float*) alloc((size_t)MBIG * DD * 4);
    float*  vc_f   = (float*) alloc((size_t)MBIG * DD * 4);
    float*  qo_f   = (float*) alloc((size_t)MBIG * DD * 4);
    bf16_t* ctx_bf = (bf16_t*)alloc((size_t)MBIG * DD * 2);
    bf16_t* Wpk[8];
    for (int i = 0; i < 8; ++i) Wpk[i] = (bf16_t*)alloc((size_t)DD * DD * 2);
    float* agent_block = (float*)alloc((size_t)6 * MAGENT * DD * 4);
    float* amean_f = agent_block + 0 * MAGENT * DD;
    float* atok_f  = agent_block + 1 * MAGENT * DD;
    float* qa_f    = agent_block + 2 * MAGENT * DD;
    float* aout_f  = agent_block + 3 * MAGENT * DD;
    float* ka_f    = agent_block + 4 * MAGENT * DD;
    float* va_f    = agent_block + 5 * MAGENT * DD;
    bf16_t* a_bf   = (bf16_t*)alloc((size_t)MAGENT * DD * 2);

    const int nBig = MBIG * DD;           // 33,554,432
    const int nAg  = MAGENT * DD;         // 131,072
    dim3 blk(256);

    // zero the padded agent buffers (determinism for padded rows)
    zero_f32_kernel<<<dim3((6 * nAg + 255) / 256), blk, 0, stream>>>(agent_block, 6 * nAg);

    // fp32 -> bf16 conversions of the big operands
    f32_to_bf16_kernel<<<dim3(nBig / 256), blk, 0, stream>>>(query, q_bf, nBig);
    f32_to_bf16_kernel<<<dim3(nBig / 256), blk, 0, stream>>>(key,   k_bf, nBig);
    f32_to_bf16_kernel<<<dim3(nBig / 256), blk, 0, stream>>>(value, v_bf, nBig);

    // pack weights to WMMA fragment order
    for (int i = 0; i < 8; ++i)
        pack_w_kernel<<<dim3((DD * DD) / 256), blk, 0, stream>>>(Wf[i], Wpk[i]);

    // agent tokens: mean -> @W_agent -> @W_qa
    agent_mean_kernel<<<dim3(nAg / 256), blk, 0, stream>>>(query, amean_f);
    f32_to_bf16_kernel<<<dim3(nAg / 256), blk, 0, stream>>>(amean_f, a_bf, nAg);
    gemm_bf16_kernel<<<dim3(DD / 64, MAGENT / 128), blk, 0, stream>>>(a_bf, Wpk[0], bf[0], atok_f, MAGENT);
    f32_to_bf16_kernel<<<dim3(nAg / 256), blk, 0, stream>>>(atok_f, a_bf, nAg);
    gemm_bf16_kernel<<<dim3(DD / 64, MAGENT / 128), blk, 0, stream>>>(a_bf, Wpk[1], bf[1], qa_f, MAGENT);

    // big projections: k_c, v_c, q_o
    gemm_bf16_kernel<<<dim3(DD / 64, MBIG / 128), blk, 0, stream>>>(k_bf, Wpk[2], bf[2], kc_f, MBIG);
    gemm_bf16_kernel<<<dim3(DD / 64, MBIG / 128), blk, 0, stream>>>(v_bf, Wpk[3], bf[3], vc_f, MBIG);
    gemm_bf16_kernel<<<dim3(DD / 64, MBIG / 128), blk, 0, stream>>>(q_bf, Wpk[4], bf[4], qo_f, MBIG);

    // agent scores -> top-k masked softmax (in-place in d_out slice)
    agent_score_kernel<<<dim3(SS / 256, BB * HH), blk, 0, stream>>>(qa_f, kc_f, attn_agent);
    topk_softmax_kernel<<<dim3(BB * HH * AA), blk, 0, stream>>>(attn_agent);

    // agent_out = attn_agent_w @ v_c
    agent_out_kernel<<<dim3(BB * HH), blk, 0, stream>>>(attn_agent, vc_f, aout_f);

    // k_a, v_a projections of agent_out
    f32_to_bf16_kernel<<<dim3(nAg / 256), blk, 0, stream>>>(aout_f, a_bf, nAg);
    gemm_bf16_kernel<<<dim3(DD / 64, MAGENT / 128), blk, 0, stream>>>(a_bf, Wpk[5], bf[5], ka_f, MAGENT);
    gemm_bf16_kernel<<<dim3(DD / 64, MAGENT / 128), blk, 0, stream>>>(a_bf, Wpk[6], bf[6], va_f, MAGENT);

    // final attention (writes attn_final and bf16 context)
    final_attn_kernel<<<dim3(NN / 256, BB * HH), blk, 0, stream>>>(qo_f, ka_f, va_f, attn_final, ctx_bf);

    // output projection
    gemm_bf16_kernel<<<dim3(DD / 64, MBIG / 128), blk, 0, stream>>>(ctx_bf, Wpk[7], bf[7], out_final, MBIG);
}
